// ExpertChoiceMoEMatcher_61323543052359
// MI455X (gfx1250) — matrix operand order repaired and checked by reading, not compiled, not run on hardware
//
#include <hip/hip_runtime.h>
#include <math.h>

typedef __attribute__((ext_vector_type(16))) __bf16 v16bf;
typedef __attribute__((ext_vector_type(8)))  __bf16 v8bf;
typedef __attribute__((ext_vector_type(8)))  float  v8f;
typedef __attribute__((ext_vector_type(4)))  unsigned int v4u;
typedef __attribute__((ext_vector_type(8)))  int    v8i;
typedef __attribute__((ext_vector_type(4)))  int    v4i;

#define BT     16384
#define DIM    512
#define NE     64
#define KNODES 512
#define FMAXF  3.402823466e38f

#if defined(__has_builtin)
#if __has_builtin(__builtin_amdgcn_tensor_load_to_lds)
#define HAVE_TDM 1
#endif
#endif
#ifndef HAVE_TDM
#define HAVE_TDM 0
#endif

// ---------------------------------------------------------------------------
// zero output accumulator + counts
// ---------------------------------------------------------------------------
__global__ void zero_kernel(float* __restrict__ out, float* __restrict__ counts) {
    size_t i = (size_t)blockIdx.x * blockDim.x + threadIdx.x;
    if (i < (size_t)BT * DIM * 2) out[i] = 0.0f;
    if (i < BT) counts[i] = 0.0f;
}

// ---------------------------------------------------------------------------
// gate scores: scores[t][e] = nan_to_num( x[t,:].flatten(1024) . gate[:,e] )
// ---------------------------------------------------------------------------
__global__ __launch_bounds__(256) void gate_kernel(const float* __restrict__ x,
                                                   const float* __restrict__ gate,
                                                   float* __restrict__ scores) {
    __shared__ float xs[4][1024];
    const int tid  = threadIdx.x;
    const int sub  = tid >> 6;
    const int lane = tid & 63;
    const int tok  = blockIdx.x * 4 + sub;

    const float4* xr  = (const float4*)(x + (size_t)tok * 1024);
    float4*       xsv = (float4*)xs[sub];
#pragma unroll
    for (int i = 0; i < 4; ++i) xsv[lane + i * 64] = xr[lane + i * 64];
    __syncthreads();

    float acc = 0.0f;
    for (int k = 0; k < 1024; ++k)
        acc = fmaf(xs[sub][k], gate[k * 64 + lane], acc);

    if (__builtin_isnan(acc)) acc = 0.0f;
    acc = fminf(fmaxf(acc, -FMAXF), FMAXF);
    scores[(size_t)tok * 64 + lane] = acc;
}

// ---------------------------------------------------------------------------
// per-expert exact top-512 (radix select, then bitonic sort)
// ---------------------------------------------------------------------------
__device__ __forceinline__ unsigned flipKey(float f) {
    unsigned u = __float_as_uint(f);
    return (u & 0x80000000u) ? ~u : (u | 0x80000000u);
}

__global__ __launch_bounds__(256) void topk_kernel(const float* __restrict__ scores,
                                                   int* __restrict__ topk_idx,
                                                   float* __restrict__ topk_scores) {
    const int e   = blockIdx.x;
    const int tid = threadIdx.x;

    __shared__ int      hist[256];
    __shared__ unsigned sPrefix;
    __shared__ int      sRemain;
    __shared__ int      sCnt;
    __shared__ float    keyS[KNODES];
    __shared__ int      keyI[KNODES];

    if (tid == 0) { sPrefix = 0u; sRemain = KNODES; }
    __syncthreads();

    for (int pass = 0; pass < 4; ++pass) {
        hist[tid] = 0;
        __syncthreads();
        const int      shift    = 24 - pass * 8;
        const unsigned prefMask = (pass == 0) ? 0u : (0xFFFFFFFFu << (shift + 8));
        const unsigned pref     = sPrefix;
        for (int t = tid; t < BT; t += 256) {
            unsigned key = flipKey(scores[(size_t)t * 64 + e]);
            if ((key & prefMask) == (pref & prefMask))
                atomicAdd(&hist[(key >> shift) & 255], 1);
        }
        __syncthreads();
        if (tid == 0) {
            int rem = sRemain;
            int b   = 255;
            for (; b > 0; --b) {
                if (hist[b] >= rem) break;
                rem -= hist[b];
            }
            sPrefix = pref | ((unsigned)b << shift);
            sRemain = rem;
        }
        __syncthreads();
    }

    const unsigned thr = sPrefix;
    if (tid == 0) sCnt = 0;
    __syncthreads();

    for (int t = tid; t < BT; t += 256) {
        float sc = scores[(size_t)t * 64 + e];
        if (flipKey(sc) > thr) {
            int p = atomicAdd(&sCnt, 1);
            keyS[p] = sc;
            keyI[p] = t;
        }
    }
    __syncthreads();
    if (tid == 0) {
        int p = sCnt;
        for (int t = 0; t < BT && p < KNODES; ++t) {
            float sc = scores[(size_t)t * 64 + e];
            if (flipKey(sc) == thr) { keyS[p] = sc; keyI[p] = t; ++p; }
        }
        sCnt = p;
    }
    __syncthreads();

    for (int size = 2; size <= KNODES; size <<= 1) {
        for (int stride = size >> 1; stride > 0; stride >>= 1) {
            __syncthreads();
            for (int i = tid; i < KNODES; i += 256) {
                int j = i ^ stride;
                if (j > i) {
                    bool  dir = ((i & size) == 0);
                    float si = keyS[i], sj = keyS[j];
                    int   ii = keyI[i], ij = keyI[j];
                    bool  jBeforeI = (sj > si) || (sj == si && ij < ii);
                    if (jBeforeI == dir) {
                        keyS[i] = sj; keyS[j] = si;
                        keyI[i] = ij; keyI[j] = ii;
                    }
                }
            }
        }
    }
    __syncthreads();
    for (int s = tid; s < KNODES; s += 256) {
        topk_idx[e * KNODES + s]    = keyI[s];
        topk_scores[e * KNODES + s] = keyS[s];
    }
}

// ---------------------------------------------------------------------------
// counts scatter (reference's transposed flattening of topk_idx)
// ---------------------------------------------------------------------------
__global__ void counts_kernel(const int* __restrict__ topk_idx, float* __restrict__ counts) {
    int i = blockIdx.x * blockDim.x + threadIdx.x;
    if (i >= NE * KNODES) return;
    int e    = i >> 9;
    int slot = i & 511;
    int dst  = topk_idx[(slot & 63) * KNODES + e * 8 + (slot >> 6)];
    atomicAdd(&counts[dst], 1.0f);
}

// ---------------------------------------------------------------------------
// vectorized fragment loads (contiguous 16B chunks -> ds_load_b128)
// ---------------------------------------------------------------------------
__device__ __forceinline__ v16bf ldFragA(const __bf16* row, int koff) {
    v8bf c0 = *(const v8bf*)(row + koff);
    v8bf c1 = *(const v8bf*)(row + koff + 16);
    return __builtin_shufflevector(c0, c1, 0, 1, 2, 3, 4, 5, 6, 7,
                                           8, 9, 10, 11, 12, 13, 14, 15);
}
__device__ __forceinline__ v16bf ldFragB(const __bf16* row, int koff) {
    v8bf c0 = *(const v8bf*)(row + koff);
    v8bf c1 = *(const v8bf*)(row + koff + 16);
    return __builtin_shufflevector(c0, c1, 0, 1, 2, 3, 4, 5, 6, 7,
                                           8, 9, 10, 11, 12, 13, 14, 15);
}

#define WMMA_BF16(A, B, C) \
    __builtin_amdgcn_wmma_f32_16x16x32_bf16(false, (A), false, (B), (short)0, (C), false, false)

#if HAVE_TDM
// issue a 2D TDM load: 32 rows x 128 f32, row stride 1024 f32
__device__ __forceinline__ void issueTdmB(const float* src, unsigned ldsAddr) {
    unsigned long long ga = (unsigned long long)(size_t)src;
    v4u g0 = {1u,                                     // count=1, user descriptor
              ldsAddr,                                // lds_addr
              (unsigned)(ga & 0xFFFFFFFFu),
              (unsigned)((ga >> 32) & 0x1FFFFFFu) | 0x80000000u};  // type=2
    v8i g1 = {(int)(2u << 16),                        // data_size=4B, wg_mask=0
              (int)(128u << 16),                      // tensor_dim0 lo16 << 16
              (int)(32u << 16),                       // tensor_dim1 lo16 << 16
              (int)(128u << 16),                      // tile_dim0 << 16
              (int)32,                                // tile_dim1
              (int)1024,                              // tensor_dim0_stride lo32
              0, 0};
    v4i gz = {0, 0, 0, 0};
#if defined(__clang_major__) && (__clang_major__ >= 23)
    v8i gz8 = {0, 0, 0, 0, 0, 0, 0, 0};
    __builtin_amdgcn_tensor_load_to_lds(g0, g1, gz, gz, gz8, 0);
#else
    __builtin_amdgcn_tensor_load_to_lds(g0, g1, gz, gz, 0);
#endif
}
#endif

// ---------------------------------------------------------------------------
// grouped complex GEMM, bf16 hi/lo split (3-term) per real product.
// Software-pipelined k-loop:
//   - TDM double-buffers the f32 B tile (issue i+1, s_wait_tensorcnt 1)
//   - A tile for i+1 register-prefetched under iteration i's WMMA block
// ---------------------------------------------------------------------------
__global__ __launch_bounds__(256) void moe_gemm_kernel(
        const float* __restrict__ x, const float* __restrict__ w,
        const int* __restrict__ topk_idx, const float* __restrict__ topk_scores,
        float* __restrict__ out) {
    const int e    = blockIdx.z;
    const int mBlk = blockIdx.y;
    const int nBlk = blockIdx.x;

    __shared__ __bf16 ArHi[64][32], ArLo[64][32], AiHi[64][32], AiLo[64][32];
    __shared__ __bf16 BrHi[64][32], BrLo[64][32], BiHi[64][32], BiLo[64][32]; // transposed (n,k)
    __shared__ float  Bf32[2][32][128];  // double-buffered TDM landing tiles
    __shared__ int    tokLds[64];
    __shared__ float  sLds[64];

    const int tid  = threadIdx.x;
    const int lane = tid & 31;
    const int wave = tid >> 5;
    const int wm   = wave & 1;
    const int wn   = wave >> 1;

    if (tid < 64) {
        int slot    = mBlk * 64 + tid;
        tokLds[tid] = topk_idx[e * KNODES + slot];
        sLds[tid]   = topk_scores[e * KNODES + slot];
    }
    __syncthreads();

    const v8f zacc = {0.f, 0.f, 0.f, 0.f, 0.f, 0.f, 0.f, 0.f};
    v8f accP[2] = {zacc, zacc}, accQ[2] = {zacc, zacc};
    v8f accR[2] = {zacc, zacc}, accS[2] = {zacc, zacc};

    // per-thread A staging coordinates: 4 threads per row, 8 k each
    const int am = tid >> 2;
    const int aq = tid & 3;
    const float4* xrow = (const float4*)(x + (size_t)tokLds[am] * (DIM * 2) + (size_t)(aq * 8) * 2);
    // per-thread B conversion coordinates
    const int bk = tid >> 3;
    const int bo = tid & 7;
    const float* wbase = w + (size_t)e * DIM * DIM * 2 + (size_t)nBlk * 128;

#if HAVE_TDM
    const unsigned ldsB0 = (unsigned)(size_t)(&Bf32[0][0][0]);
    const unsigned ldsB1 = (unsigned)(size_t)(&Bf32[1][0][0]);
    if (wave == 0) issueTdmB(wbase, ldsB0);          // prologue: TDM for k-step 0
#endif
    float4 xa[4];                                     // A register prefetch (k-step 0)
#pragma unroll
    for (int v = 0; v < 4; ++v) xa[v] = xrow[v];

    for (int it = 0; it < DIM / 32; ++it) {
        const int k0 = it * 32;

#if HAVE_TDM
        if (it + 1 < DIM / 32) {
            if (wave == 0)
                issueTdmB(wbase + (size_t)(k0 + 32) * (DIM * 2), (it & 1) ? ldsB0 : ldsB1);
            __builtin_amdgcn_s_wait_tensorcnt(1);     // buffer `it` has landed (in-order)
        } else {
            __builtin_amdgcn_s_wait_tensorcnt(0);
        }
#endif
        __syncthreads();   // prev iteration's fragment reads are done; LDS writable

        // ---- convert prefetched A regs -> bf16 hi/lo planes ----
        {
            float vbuf[16];
#pragma unroll
            for (int v = 0; v < 4; ++v) {
                vbuf[v * 4 + 0] = xa[v].x; vbuf[v * 4 + 1] = xa[v].y;
                vbuf[v * 4 + 2] = xa[v].z; vbuf[v * 4 + 3] = xa[v].w;
            }
            v8bf rh, rl, ih, il;
#pragma unroll
            for (int u = 0; u < 8; ++u) {
                float vr = vbuf[2 * u], vi = vbuf[2 * u + 1];
                __bf16 h0 = (__bf16)vr, h1 = (__bf16)vi;
                rh[u] = h0; rl[u] = (__bf16)(vr - (float)h0);
                ih[u] = h1; il[u] = (__bf16)(vi - (float)h1);
            }
            *(v8bf*)&ArHi[am][aq * 8] = rh;
            *(v8bf*)&ArLo[am][aq * 8] = rl;
            *(v8bf*)&AiHi[am][aq * 8] = ih;
            *(v8bf*)&AiLo[am][aq * 8] = il;
        }

        // ---- convert B f32 tile -> transposed bf16 hi/lo planes ----
        {
#if HAVE_TDM
            const float* bp = &Bf32[it & 1][bk][bo * 16];
#else
            const float* bp = wbase + (size_t)(k0 + bk) * (DIM * 2) + bo * 16;
#endif
#pragma unroll
            for (int u = 0; u < 8; ++u) {
                float vr = bp[2 * u], vi = bp[2 * u + 1];
                __bf16 h0 = (__bf16)vr, h1 = (__bf16)vi;
                int n = bo * 8 + u;
                BrHi[n][bk] = h0; BrLo[n][bk] = (__bf16)(vr - (float)h0);
                BiHi[n][bk] = h1; BiLo[n][bk] = (__bf16)(vi - (float)h1);
            }
        }
        __syncthreads();

        // ---- register-prefetch next iteration's A tile (hides under WMMAs) ----
        if (it + 1 < DIM / 32) {
            const float4* xp = xrow + (size_t)(k0 + 32) * 2 / 4;
#pragma unroll
            for (int v = 0; v < 4; ++v) xa[v] = xp[v];
        }

        // ---- fragments + 24 WMMAs per wave per k-step ----
        const int nRow = wn * 16 + (lane & 15);
        const int kB   = (lane >> 4) * 8;
        const v16bf brh = ldFragB(&BrHi[nRow][0], kB);
        const v16bf brl = ldFragB(&BrLo[nRow][0], kB);
        const v16bf bih = ldFragB(&BiHi[nRow][0], kB);
        const v16bf bil = ldFragB(&BiLo[nRow][0], kB);

#pragma unroll
        for (int mt = 0; mt < 2; ++mt) {
            const int mRow = wm * 32 + mt * 16 + (lane & 15);
            const int kA   = (lane >> 4) * 8;
            const v16bf arh = ldFragA(&ArHi[mRow][0], kA);
            const v16bf arl = ldFragA(&ArLo[mRow][0], kA);
            const v16bf aih = ldFragA(&AiHi[mRow][0], kA);
            const v16bf ail = ldFragA(&AiLo[mRow][0], kA);

            accP[mt] = WMMA_BF16(arh, brh, accP[mt]);
            accP[mt] = WMMA_BF16(arh, brl, accP[mt]);
            accP[mt] = WMMA_BF16(arl, brh, accP[mt]);

            accQ[mt] = WMMA_BF16(aih, bih, accQ[mt]);
            accQ[mt] = WMMA_BF16(aih, bil, accQ[mt]);
            accQ[mt] = WMMA_BF16(ail, bih, accQ[mt]);

            accR[mt] = WMMA_BF16(arh, bih, accR[mt]);
            accR[mt] = WMMA_BF16(arh, bil, accR[mt]);
            accR[mt] = WMMA_BF16(arl, bih, accR[mt]);

            accS[mt] = WMMA_BF16(aih, brh, accS[mt]);
            accS[mt] = WMMA_BF16(aih, brl, accS[mt]);
            accS[mt] = WMMA_BF16(ail, brh, accS[mt]);
        }
    }
    __syncthreads();

    // ---- epilogue: weight by routing score, scatter-add (transposed index) ----
    const int nGlob = nBlk * 64 + wn * 16 + (lane & 15);
#pragma unroll
    for (int mt = 0; mt < 2; ++mt) {
#pragma unroll
        for (int r = 0; r < 8; ++r) {
            const int mLoc = wm * 32 + mt * 16 + r + ((lane >> 4) << 3);
            const int slot = mBlk * 64 + mLoc;
            const float sc = sLds[mLoc];
            const int dst  = topk_idx[(slot & 63) * KNODES + e * 8 + (slot >> 6)];
            float yr = accP[mt][r] - accQ[mt][r];
            float yi = accR[mt][r] + accS[mt][r];
            yr = __builtin_isnan(yr) ? 0.0f : fminf(fmaxf(yr, -FMAXF), FMAXF);
            yi = __builtin_isnan(yi) ? 0.0f : fminf(fmaxf(yi, -FMAXF), FMAXF);
            float* op = out + (size_t)dst * (DIM * 2) + (size_t)nGlob * 2;
            atomicAdd(op,     yr * sc);
            atomicAdd(op + 1, yi * sc);
        }
    }
}

// ---------------------------------------------------------------------------
// finalize: agg = out / max(counts,1);  gelu(agg + bias) exact (erf)
// ---------------------------------------------------------------------------
__global__ void finalize_kernel(float* __restrict__ out,
                                const float* __restrict__ counts,
                                const float* __restrict__ bias) {
    size_t i = (size_t)blockIdx.x * blockDim.x + threadIdx.x;
    if (i >= (size_t)BT * DIM) return;
    const int t = (int)(i >> 9);
    const int d = (int)(i & 511);
    const float inv = 1.0f / fmaxf(counts[t], 1.0f);
    const float b   = bias[d];
    float* p = out + i * 2;
#pragma unroll
    for (int c = 0; c < 2; ++c) {
        float v = p[c] * inv + b;
        p[c] = 0.5f * v * (1.0f + erff(v * 0.7071067811865475f));
    }
}

// ---------------------------------------------------------------------------
extern "C" void kernel_launch(void* const* d_in, const int* in_sizes, int n_in,
                              void* d_out, int out_size, void* d_ws, size_t ws_size,
                              hipStream_t stream) {
    (void)in_sizes; (void)n_in; (void)out_size; (void)ws_size;
    const float* x    = (const float*)d_in[0];  // [16384,512,2]
    const float* gate = (const float*)d_in[1];  // [1024,64]
    const float* w    = (const float*)d_in[2];  // [64,512,512,2]
    const float* bias = (const float*)d_in[3];  // [512]
    float* out = (float*)d_out;                 // [16384,512,2]

    char*  ws          = (char*)d_ws;
    float* scores      = (float*)ws;                      // 16384*64 f32
    int*   topk_idx    = (int*)(ws + 4194304);            // 64*512 i32
    float* topk_scores = (float*)(ws + 4194304 + 131072); // 64*512 f32
    float* counts      = (float*)(ws + 4194304 + 262144); // 16384 f32

    zero_kernel<<<(BT * DIM * 2 + 255) / 256, 256, 0, stream>>>(out, counts);
    gate_kernel<<<BT / 4, 256, 0, stream>>>(x, gate, scores);
    topk_kernel<<<NE, 256, 0, stream>>>(scores, topk_idx, topk_scores);
    counts_kernel<<<(NE * KNODES + 255) / 256, 256, 0, stream>>>(topk_idx, counts);
    dim3 gemmGrid(8, 8, NE);
    moe_gemm_kernel<<<gemmGrid, 256, 0, stream>>>(x, w, topk_idx, topk_scores, out);
    finalize_kernel<<<((size_t)BT * DIM + 255) / 256, 256, 0, stream>>>(out, counts, bias);
}
